// multiHeadAttention_14121852470086
// MI455X (gfx1250) — compile-verified, matrix operand order checked by reading, not run on hardware
//
#include <hip/hip_runtime.h>

#define EMB    1024
#define NSEQ   2048
#define NBATCH 2
#define HEADS  16
#define HDIM   64
#define MTOT   (NBATCH * NSEQ)   // 4096

typedef __attribute__((ext_vector_type(16))) __bf16 v16bf;
typedef __attribute__((ext_vector_type(8)))  float  v8f;

union Frag { v16bf v; uint4 u[2]; };

__device__ __forceinline__ unsigned short f2b(float f) {
    union { __bf16 h; unsigned short s; } c;
    c.h = (__bf16)f;                       // hardware f32->bf16 convert
    return c.s;
}
__device__ __forceinline__ unsigned int pk2(float a, float b) {
    union { __bf16 h[2]; unsigned int u; } c;
    c.h[0] = (__bf16)a; c.h[1] = (__bf16)b;
    return c.u;
}

// async copy 16B global -> LDS, tracked by ASYNCcnt (no VGPR round trip)
__device__ __forceinline__ void async_cp16(unsigned lds_off, const void* gptr) {
    unsigned long long ga = (unsigned long long)(uintptr_t)gptr;
    asm volatile("global_load_async_to_lds_b128 %0, %1, off"
                 :: "v"(lds_off), "v"(ga) : "memory");
}
__device__ __forceinline__ void wait_async() {
    asm volatile("s_wait_asynccnt 0" ::: "memory");
}
__device__ __forceinline__ unsigned lds_off_of(const void* p) {
    return (unsigned)(uintptr_t)p;         // low 32 bits of flat addr = LDS offset
}

// butterfly max over the 16-lane DPP row (rows 0-7 live in lanes 0-15,
// rows 8-15 in lanes 16-31; xor masks 1/2/4/8 stay inside the row).
// VALU-only -> co-executes with WMMA, no LDS pipe, no dscnt waits.
__device__ __forceinline__ float rowmax16(float x) {
    int v = __float_as_int(x), t;
    t = __builtin_amdgcn_update_dpp(0, v, 0x161, 0xf, 0xf, true);   // row_xmask:1
    v = __float_as_int(fmaxf(__int_as_float(v), __int_as_float(t)));
    t = __builtin_amdgcn_update_dpp(0, v, 0x162, 0xf, 0xf, true);   // row_xmask:2
    v = __float_as_int(fmaxf(__int_as_float(v), __int_as_float(t)));
    t = __builtin_amdgcn_update_dpp(0, v, 0x164, 0xf, 0xf, true);   // row_xmask:4
    v = __float_as_int(fmaxf(__int_as_float(v), __int_as_float(t)));
    t = __builtin_amdgcn_update_dpp(0, v, 0x168, 0xf, 0xf, true);   // row_xmask:8
    v = __float_as_int(fmaxf(__int_as_float(v), __int_as_float(t)));
    return __int_as_float(v);
}

// ---------------------------------------------------------------------------
// Kernel 1: QKV projection.  out = X(4096x1024) @ W(1024x1024) + b, bf16 out.
// MODE 0: [bh][n][d] (Q, K).  MODE 1: [bh][d][n] (V transposed).
// ---------------------------------------------------------------------------
template <int MODE>
__global__ __launch_bounds__(256)
void proj_qkv_kernel(const float* __restrict__ X, const float* __restrict__ W,
                     const float* __restrict__ bias, unsigned short* __restrict__ out)
{
    __shared__ unsigned short As[128 * 32];   // row-major [m][k] bf16
    __shared__ unsigned short Bs[64 * 32];    // transposed [n][k] bf16

    const int tid  = threadIdx.x;
    const int lane = tid & 31;
    const int w    = tid >> 5;
    const int lm   = lane & 15;
    const int hi   = lane >> 4;
    const int n0   = blockIdx.x * 64;
    const int m0   = blockIdx.y * 128;

    v8f acc[4] = {};

    for (int k0 = 0; k0 < EMB; k0 += 32) {
        {   // stage A tile (128x32 f32 -> bf16), 16 elems / thread
            const int row = tid >> 1;
            const int kh  = (tid & 1) * 16;
            const float4* s4 = (const float4*)(X + (size_t)(m0 + row) * EMB + k0 + kh);
            float4 f0 = s4[0], f1 = s4[1], f2 = s4[2], f3 = s4[3];
            uint4 q0 = make_uint4(pk2(f0.x, f0.y), pk2(f0.z, f0.w),
                                  pk2(f1.x, f1.y), pk2(f1.z, f1.w));
            uint4 q1 = make_uint4(pk2(f2.x, f2.y), pk2(f2.z, f2.w),
                                  pk2(f3.x, f3.y), pk2(f3.z, f3.w));
            *(uint4*)&As[row * 32 + kh]     = q0;
            *(uint4*)&As[row * 32 + kh + 8] = q1;
        }
        {   // stage B tile (W rows k0..k0+31, cols n0..n0+63), transposed
            const int kr = tid >> 3;
            const int nc = (tid & 7) * 8;
            const float4* t4 = (const float4*)(W + (size_t)(k0 + kr) * EMB + n0 + nc);
            float4 g0 = t4[0], g1 = t4[1];
            Bs[(nc + 0) * 32 + kr] = f2b(g0.x);
            Bs[(nc + 1) * 32 + kr] = f2b(g0.y);
            Bs[(nc + 2) * 32 + kr] = f2b(g0.z);
            Bs[(nc + 3) * 32 + kr] = f2b(g0.w);
            Bs[(nc + 4) * 32 + kr] = f2b(g1.x);
            Bs[(nc + 5) * 32 + kr] = f2b(g1.y);
            Bs[(nc + 6) * 32 + kr] = f2b(g1.z);
            Bs[(nc + 7) * 32 + kr] = f2b(g1.w);
        }
        __syncthreads();

        Frag fa;
        const unsigned short* arow = &As[(w * 16 + lm) * 32];
        fa.u[0] = *(const uint4*)(arow + hi * 8);
        fa.u[1] = *(const uint4*)(arow + 16 + hi * 8);
#pragma unroll
        for (int ns = 0; ns < 4; ++ns) {
            Frag fb;
            const unsigned short* brow = &Bs[(ns * 16 + lm) * 32];
            fb.u[0] = *(const uint4*)(brow + hi * 8);
            fb.u[1] = *(const uint4*)(brow + 16 + hi * 8);
            acc[ns] = __builtin_amdgcn_wmma_f32_16x16x32_bf16(
                false, fa.v, false, fb.v, (short)0, acc[ns], false, false);
        }
        __syncthreads();
    }

#pragma unroll
    for (int ns = 0; ns < 4; ++ns) {
        const int n = n0 + ns * 16 + lm;
        const int h = n >> 6;
        const int d = n & 63;
        const float bv = bias[n];
#pragma unroll
        for (int r = 0; r < 8; ++r) {
            const int m  = m0 + w * 16 + hi * 8 + r;
            const int b  = m >> 11;
            const int nq = m & 2047;
            const float val = acc[ns][r] + bv;
            size_t idx;
            if (MODE == 0) idx = ((size_t)(b * HEADS + h) * NSEQ + nq) * HDIM + d;
            else           idx = ((size_t)(b * HEADS + h) * HDIM + d) * NSEQ + nq;
            out[idx] = f2b(val);
        }
    }
}

// ---------------------------------------------------------------------------
// Kernel 2: flash attention with double-buffered async K/V panels.
// 8 waves x 16 query rows = 128 queries / block.  Panel i+1 is DMA'd to LDS
// (ASYNCcnt) while panel i is consumed by WMMAs.  Row-max via DPP butterfly,
// row-sum via one WMMA against an all-ones B.  Final scale folds the
// reference's post-softmax /sqrt(EMB)=32.
// ---------------------------------------------------------------------------
__global__ __launch_bounds__(256)
void attn_kernel(const unsigned short* __restrict__ Q,
                 const unsigned short* __restrict__ K,
                 const unsigned short* __restrict__ VT,
                 unsigned short* __restrict__ ctx)
{
    __shared__ unsigned short Ks[2][32 * 64];    // [key][d] bf16, 2 x 4KB
    __shared__ unsigned short Vs[2][64 * 32];    // [d][key] bf16, 2 x 4KB
    __shared__ unsigned short Pl[8 * 16 * 32];   // per-wave P slab, 8KB

    const int tid  = threadIdx.x;
    const int lane = tid & 31;
    const int w    = tid >> 5;
    const int lm   = lane & 15;
    const int hi   = lane >> 4;
    const int bh   = blockIdx.y;
    const int q0   = blockIdx.x * 128 + w * 16;

    // staging coordinates (per thread)
    const int skey = tid >> 3, sdp = (tid & 7) * 8;   // K panel: 32x64
    const int svd  = tid >> 2, skp = (tid & 3) * 8;   // V panel: 64x32
    const unsigned short* kgb = K  + ((size_t)bh * NSEQ + skey) * HDIM + sdp;
    const unsigned short* vgb = VT + ((size_t)bh * HDIM + svd) * NSEQ + skp;

    // Q fragments (d halves) pinned in VGPRs for the whole kernel.
    Frag qa[2];
    {
        const unsigned short* qrow = Q + ((size_t)bh * NSEQ + q0 + lm) * HDIM;
#pragma unroll
        for (int hf = 0; hf < 2; ++hf) {
            qa[hf].u[0] = *(const uint4*)(qrow + hf * 32 + hi * 8);
            qa[hf].u[1] = *(const uint4*)(qrow + hf * 32 + 16 + hi * 8);
        }
    }

    // all-ones bf16 B fragment for the row-sum WMMA
    Frag fones;
#pragma unroll
    for (int i = 0; i < 2; ++i)
        fones.u[i] = make_uint4(0x3F803F80u, 0x3F803F80u, 0x3F803F80u, 0x3F803F80u);

    v8f o0 = {}, o1 = {}, o2 = {}, o3 = {};
    float mrun[8], lrun[8];
#pragma unroll
    for (int r = 0; r < 8; ++r) { mrun[r] = -1e30f; lrun[r] = 0.0f; }

    unsigned short* pw = &Pl[w * 16 * 32];

    // prologue: stage panel 0
    async_cp16(lds_off_of(&Ks[0][skey * 64 + sdp]), kgb);
    async_cp16(lds_off_of(&Vs[0][svd * 32 + skp]), vgb);
    wait_async();
    __syncthreads();

    int buf = 0;
    for (int kb = 0; kb < NSEQ; kb += 32) {
        // ---- kick off DMA of panel i+1 into the other buffer ----
        if (kb + 32 < NSEQ) {
            async_cp16(lds_off_of(&Ks[buf ^ 1][skey * 64 + sdp]),
                       kgb + (size_t)(kb + 32) * HDIM);
            async_cp16(lds_off_of(&Vs[buf ^ 1][svd * 32 + skp]),
                       vgb + (kb + 32));
            if (kb + 64 < NSEQ)                    // prefetch panel i+2
                __builtin_prefetch(kgb + (size_t)(kb + 64) * HDIM, 0, 1);
        }

        // ---- preload all K fragments, then batch the score WMMAs ----
        Frag fk[2][2];
#pragma unroll
        for (int kt = 0; kt < 2; ++kt) {
            const unsigned short* krow = &Ks[buf][(kt * 16 + lm) * 64];
#pragma unroll
            for (int hf = 0; hf < 2; ++hf) {
                fk[kt][hf].u[0] = *(const uint4*)(krow + hf * 32 + hi * 8);
                fk[kt][hf].u[1] = *(const uint4*)(krow + hf * 32 + 16 + hi * 8);
            }
        }
        v8f s0 = {}, s1 = {};
        s0 = __builtin_amdgcn_wmma_f32_16x16x32_bf16(
            false, qa[0].v, false, fk[0][0].v, (short)0, s0, false, false);
        s1 = __builtin_amdgcn_wmma_f32_16x16x32_bf16(
            false, qa[0].v, false, fk[1][0].v, (short)0, s1, false, false);
        s0 = __builtin_amdgcn_wmma_f32_16x16x32_bf16(
            false, qa[1].v, false, fk[0][1].v, (short)0, s0, false, false);
        s1 = __builtin_amdgcn_wmma_f32_16x16x32_bf16(
            false, qa[1].v, false, fk[1][1].v, (short)0, s1, false, false);

        // ---- online softmax: DPP row-max butterfly, exp, rescale ----
        float p0[8], p1[8], sc[8];
#pragma unroll
        for (int r = 0; r < 8; ++r) {
            float a = s0[r], b = s1[r];
            const float rmax = rowmax16(fmaxf(a, b));
            const float mnew = fmaxf(mrun[r], rmax);
            sc[r]  = __expf(mrun[r] - mnew);
            p0[r]  = __expf(a - mnew);
            p1[r]  = __expf(b - mnew);
            mrun[r] = mnew;
            o0[r] *= sc[r]; o1[r] *= sc[r]; o2[r] *= sc[r]; o3[r] *= sc[r];
        }

        // ---- reshape P (C/D layout) -> A-fragment layout via wave-private LDS
#pragma unroll
        for (int r = 0; r < 8; ++r) {
            pw[(hi * 8 + r) * 32 + lm]      = f2b(p0[r]);
            pw[(hi * 8 + r) * 32 + 16 + lm] = f2b(p1[r]);
        }
        asm volatile("s_wait_dscnt 0" ::: "memory");   // cross-lane RAW inside wave
        Frag pa;
        pa.u[0] = *(const uint4*)(pw + lm * 32 + hi * 8);
        pa.u[1] = *(const uint4*)(pw + lm * 32 + 16 + hi * 8);

        // ---- preload V fragments, then batch row-sum + PV WMMAs ----
        Frag fv[4];
#pragma unroll
        for (int dt = 0; dt < 4; ++dt) {
            const unsigned short* vrow = &Vs[buf][(dt * 16 + lm) * 32];
            fv[dt].u[0] = *(const uint4*)(vrow + hi * 8);
            fv[dt].u[1] = *(const uint4*)(vrow + 16 + hi * 8);
        }
        v8f z = {};
        v8f lsum = __builtin_amdgcn_wmma_f32_16x16x32_bf16(
            false, pa.v, false, fones.v, (short)0, z, false, false);
        o0 = __builtin_amdgcn_wmma_f32_16x16x32_bf16(
            false, pa.v, false, fv[0].v, (short)0, o0, false, false);
        o1 = __builtin_amdgcn_wmma_f32_16x16x32_bf16(
            false, pa.v, false, fv[1].v, (short)0, o1, false, false);
        o2 = __builtin_amdgcn_wmma_f32_16x16x32_bf16(
            false, pa.v, false, fv[2].v, (short)0, o2, false, false);
        o3 = __builtin_amdgcn_wmma_f32_16x16x32_bf16(
            false, pa.v, false, fv[3].v, (short)0, o3, false, false);
#pragma unroll
        for (int r = 0; r < 8; ++r)
            lrun[r] = lrun[r] * sc[r] + lsum[r];

        // ---- pipeline boundary: my DMA landed + everyone done with buf ----
        wait_async();
        __syncthreads();
        buf ^= 1;
    }

    // ---- epilogue: 1/l and the reference's post-softmax /sqrt(EMB) ----
    const int b = bh / HEADS, h = bh % HEADS;
#pragma unroll
    for (int r = 0; r < 8; ++r) {
        const float inv = (1.0f / 32.0f) / lrun[r];
        const int nq = q0 + hi * 8 + r;
        unsigned short* crow = ctx + ((size_t)(b * NSEQ + nq)) * EMB + h * HDIM;
        crow[0 * 16 + lm] = f2b(o0[r] * inv);
        crow[1 * 16 + lm] = f2b(o1[r] * inv);
        crow[2 * 16 + lm] = f2b(o2[r] * inv);
        crow[3 * 16 + lm] = f2b(o3[r] * inv);
    }
}

// ---------------------------------------------------------------------------
// Kernel 3: output projection.  out(f32) = ctx(bf16) @ Wo + bo.
// A tile is already bf16 -> async global->LDS copy, no VGPR round trip.
// ---------------------------------------------------------------------------
__global__ __launch_bounds__(256)
void proj_out_kernel(const unsigned short* __restrict__ A, const float* __restrict__ W,
                     const float* __restrict__ bias, float* __restrict__ out)
{
    __shared__ unsigned short As[128 * 32];
    __shared__ unsigned short Bs[64 * 32];

    const int tid  = threadIdx.x;
    const int lane = tid & 31;
    const int w    = tid >> 5;
    const int lm   = lane & 15;
    const int hi   = lane >> 4;
    const int n0   = blockIdx.x * 64;
    const int m0   = blockIdx.y * 128;

    v8f acc[4] = {};

    for (int k0 = 0; k0 < EMB; k0 += 32) {
        {   // A tile: bf16 copy via async-to-LDS (2 x 16B per thread)
            const int row = tid >> 1;
            const int kh  = (tid & 1) * 16;
            const unsigned short* src = A + (size_t)(m0 + row) * EMB + k0 + kh;
            async_cp16(lds_off_of(&As[row * 32 + kh]),     src);
            async_cp16(lds_off_of(&As[row * 32 + kh + 8]), src + 8);
        }
        {   // B tile: f32 -> bf16, transposed
            const int kr = tid >> 3;
            const int nc = (tid & 7) * 8;
            const float4* t4 = (const float4*)(W + (size_t)(k0 + kr) * EMB + n0 + nc);
            float4 g0 = t4[0], g1 = t4[1];
            Bs[(nc + 0) * 32 + kr] = f2b(g0.x);
            Bs[(nc + 1) * 32 + kr] = f2b(g0.y);
            Bs[(nc + 2) * 32 + kr] = f2b(g0.z);
            Bs[(nc + 3) * 32 + kr] = f2b(g0.w);
            Bs[(nc + 4) * 32 + kr] = f2b(g1.x);
            Bs[(nc + 5) * 32 + kr] = f2b(g1.y);
            Bs[(nc + 6) * 32 + kr] = f2b(g1.z);
            Bs[(nc + 7) * 32 + kr] = f2b(g1.w);
        }
        wait_async();
        __syncthreads();

        Frag fa;
        const unsigned short* arow = &As[(w * 16 + lm) * 32];
        fa.u[0] = *(const uint4*)(arow + hi * 8);
        fa.u[1] = *(const uint4*)(arow + 16 + hi * 8);
#pragma unroll
        for (int ns = 0; ns < 4; ++ns) {
            Frag fb;
            const unsigned short* brow = &Bs[(ns * 16 + lm) * 32];
            fb.u[0] = *(const uint4*)(brow + hi * 8);
            fb.u[1] = *(const uint4*)(brow + 16 + hi * 8);
            acc[ns] = __builtin_amdgcn_wmma_f32_16x16x32_bf16(
                false, fa.v, false, fb.v, (short)0, acc[ns], false, false);
        }
        __syncthreads();
    }

#pragma unroll
    for (int ns = 0; ns < 4; ++ns) {
        const int n = n0 + ns * 16 + lm;
        const float bv = bias[n];
#pragma unroll
        for (int r = 0; r < 8; ++r) {
            const int m = m0 + w * 16 + hi * 8 + r;
            out[(size_t)m * EMB + n] = acc[ns][r] + bv;
        }
    }
}

// ---------------------------------------------------------------------------
extern "C" void kernel_launch(void* const* d_in, const int* in_sizes, int n_in,
                              void* d_out, int out_size, void* d_ws, size_t ws_size,
                              hipStream_t stream)
{
    const float* x  = (const float*)d_in[0];
    const float* Wq = (const float*)d_in[1];
    const float* bq = (const float*)d_in[2];
    const float* Wk = (const float*)d_in[3];
    const float* bk = (const float*)d_in[4];
    const float* Wv = (const float*)d_in[5];
    const float* bv = (const float*)d_in[6];
    const float* Wo = (const float*)d_in[7];
    const float* bo = (const float*)d_in[8];
    float* out = (float*)d_out;

    const size_t elems = (size_t)MTOT * EMB;
    unsigned short* qb = (unsigned short*)d_ws;
    unsigned short* kb = qb + elems;
    unsigned short* vb = kb + elems;
    unsigned short* cb = vb + elems;

    dim3 blk(256);
    dim3 gproj(EMB / 64, MTOT / 128);            // (16, 32)
    dim3 gattn(NSEQ / 128, NBATCH * HEADS);      // (16, 32)

    proj_qkv_kernel<0><<<gproj, blk, 0, stream>>>(x, Wq, bq, qb);
    proj_qkv_kernel<0><<<gproj, blk, 0, stream>>>(x, Wk, bk, kb);
    proj_qkv_kernel<1><<<gproj, blk, 0, stream>>>(x, Wv, bv, vb);
    attn_kernel<<<gattn, blk, 0, stream>>>(qb, kb, vb, cb);
    proj_out_kernel<<<gproj, blk, 0, stream>>>(cb, Wo, bo, out);
}
